// TransformerXL_22840636081002
// MI455X (gfx1250) — matrix operand order repaired and checked by reading, not compiled, not run on hardware
//
#include <hip/hip_runtime.h>

#define QLEN   1024
#define BSZ    4
#define NT     (QLEN*BSZ)     // 4096 tokens
#define DM     1024
#define NH     16
#define DHD    64
#define FF     4096
#define OUTD   1024
#define NL     4
#define KL     1024
#define MEMLEN 512

typedef __bf16 bf16;
typedef __attribute__((ext_vector_type(16))) __bf16 v16bf;
typedef __attribute__((ext_vector_type(8)))  float  v8f;

// ---------------- WMMA fragment helpers (contiguous k = 16*(lane/16) + s in both A and B) -------------

// A-matrix 16x32 bf16, row-major source (p -> element [0][0]), row stride ld.
static __device__ inline v16bf ldA(const bf16* p, int ld) {
  const int lane = threadIdx.x & 31;
  const int lh = lane >> 4, m = lane & 15;
  const bf16* row = p + (size_t)m * ld;
  v16bf a;
#pragma unroll
  for (int r = 0; r < 8; ++r) {
    int kb = 16*lh + 2*r;
    a[2*r]   = row[kb];
    a[2*r+1] = row[kb+1];
  }
  return a;
}

// A fragment with row clamping (base -> matrix row 0), k0 = K offset.
static __device__ inline v16bf ldA_clamp(const bf16* base, int ld, int row0, int rmax, int k0) {
  const int lane = threadIdx.x & 31;
  const int lh = lane >> 4, m = lane & 15;
  int row = row0 + m; if (row > rmax) row = rmax;
  const bf16* p = base + (size_t)row * ld + k0;
  v16bf a;
#pragma unroll
  for (int r = 0; r < 8; ++r) {
    int kb = 16*lh + 2*r;
    a[2*r]   = p[kb];
    a[2*r+1] = p[kb+1];
  }
  return a;
}

// B-matrix 32x16 bf16 from K x N row-major storage (s -> element [0][0]).
static __device__ inline v16bf ldB(const bf16* s, int ld) {
  const int lane = threadIdx.x & 31;
  const int lh = lane >> 4, n = lane & 15;
  v16bf b;
#pragma unroll
  for (int r = 0; r < 8; ++r) {
    int k = 16*lh + 2*r;
    b[2*r]   = s[(size_t)k * ld + n];
    b[2*r+1] = s[(size_t)(k+1) * ld + n];
  }
  return b;
}

// B-matrix 32x16 bf16 from N x K row-major storage (weights W[n][k]).
static __device__ inline v16bf ldBT(const bf16* w, int ld) {
  const int lane = threadIdx.x & 31;
  const int lh = lane >> 4, n = lane & 15;
  const bf16* p = w + (size_t)n * ld + 16*lh;
  v16bf b;
#pragma unroll
  for (int r = 0; r < 8; ++r) {
    b[2*r]   = p[2*r];
    b[2*r+1] = p[2*r+1];
  }
  return b;
}

static __device__ inline v8f wmma_bf16(v16bf a, v16bf b, v8f c) {
  return __builtin_amdgcn_wmma_f32_16x16x32_bf16(false, a, false, b, (short)0, c, false, false);
}

// ---------------- generic GEMM: C[M,N] = A[M,K](bf16) * W[N,K]^T + bias -------
// block tile 128(M) x 128(N), K-step 32; 8 waves, wave = 16 rows x 128 cols.

__global__ __launch_bounds__(256) void gemm_kernel(
    const bf16* __restrict__ A, const bf16* __restrict__ W,
    const float* __restrict__ bias, float* __restrict__ Cf,
    bf16* __restrict__ Cb, int M, int N, int K, int relu)
{
  __shared__ bf16 As[128*40];
  __shared__ bf16 Ws[128*40];
  const int m0 = blockIdx.y * 128;
  const int n0 = blockIdx.x * 128;
  const int wv = threadIdx.x >> 5;
  const int lane = threadIdx.x & 31;
  const int lh = lane >> 4, nl = lane & 15;

  // each thread stages 2 A chunks + 2 W chunks per K-step (int4 = 8 bf16)
  const int c0 = threadIdx.x, c1 = threadIdx.x + 256;
  const int ar0 = c0 >> 2, ac0 = (c0 & 3) * 8;
  const int ar1 = c1 >> 2, ac1 = (c1 & 3) * 8;

  v8f acc[8] = {};

  for (int k0 = 0; k0 < K; k0 += 32) {
    // stage to registers first: 4 independent b128 loads in flight
    int4 a0 = *(const int4*)(A + (size_t)(m0 + ar0)*K + k0 + ac0);
    int4 a1 = *(const int4*)(A + (size_t)(m0 + ar1)*K + k0 + ac1);
    int4 w0 = *(const int4*)(W + (size_t)(n0 + ar0)*K + k0 + ac0);
    int4 w1 = *(const int4*)(W + (size_t)(n0 + ar1)*K + k0 + ac1);
    if (k0 + 32 < K) {
      __builtin_prefetch(A + (size_t)(m0 + ar0)*K + k0 + 32 + ac0, 0, 1);
      __builtin_prefetch(W + (size_t)(n0 + ar0)*K + k0 + 32 + ac0, 0, 1);
    }
    __syncthreads();   // previous iteration's LDS reads complete
    *(int4*)(As + ar0*40 + ac0) = a0;
    *(int4*)(As + ar1*40 + ac1) = a1;
    *(int4*)(Ws + ar0*40 + ac0) = w0;
    *(int4*)(Ws + ar1*40 + ac1) = w1;
    __syncthreads();
    v16bf a = ldA(As + wv*16*40, 40);
#pragma unroll
    for (int nt = 0; nt < 8; ++nt)
      acc[nt] = wmma_bf16(a, ldBT(Ws + nt*16*40, 40), acc[nt]);
  }

#pragma unroll
  for (int nt = 0; nt < 8; ++nt) {
    int col = n0 + nt*16 + nl;
    float bs = bias ? bias[col] : 0.f;
#pragma unroll
    for (int r = 0; r < 8; ++r) {
      int row = m0 + wv*16 + r + 8*lh;
      float v = acc[nt][r] + bs;
      if (relu) v = fmaxf(v, 0.f);
      size_t o = (size_t)row * N + col;
      if (Cf) Cf[o] = v;
      if (Cb) Cb[o] = (bf16)v;
    }
  }
}

// ---------------- fused relative attention (per b,h / 16 q-rows) -------------

__global__ __launch_bounds__(256) void attn_kernel(
    const bf16* __restrict__ qu, const bf16* __restrict__ qv,
    const bf16* __restrict__ kk, const bf16* __restrict__ vvv,
    const bf16* __restrict__ rh, float* __restrict__ av)
{
  extern __shared__ char smem[];
  float* raw  = (float*)smem;              // 32 x 1024 f32 = 128KB
  bf16*  pbuf = (bf16*)smem;               // reuse: 16 x 1024 bf16 (32KB)
  float* part = (float*)(smem + 98304);    // reuse: 2 x 16 x 64 f32 (8KB)
  float* sc   = (float*)(smem + 131072);   // 16 x 1024 f32 = 64KB

  const int i0 = blockIdx.x * 16;
  const int bh = blockIdx.y;
  const int b = bh / NH, h = bh % NH;
  const int wv = threadIdx.x >> 5;
  const int lane = threadIdx.x & 31;
  const int lh = lane >> 4, nl = lane & 15;

  // ---- phase 1: raw[r][j] = (q+v)[i0+r] . r_head[j]   (32 rows, clamped) ----
  {
    const bf16* qvb = qv + (size_t)bh * QLEN * DHD;
    const bf16* rhb = rh + (size_t)h * KL * DHD;
    int rowsub = wv & 1;
    int row0 = i0 + rowsub * 16;
    v16bf a0 = ldA_clamp(qvb, DHD, row0, QLEN-1, 0);
    v16bf a1 = ldA_clamp(qvb, DHD, row0, QLEN-1, 32);
    for (int c = 0; c < 16; ++c) {
      int j0 = ((wv >> 1) * 16 + c) * 16;
      v8f acc = {};
      acc = wmma_bf16(a0, ldBT(rhb + (size_t)j0*DHD, DHD), acc);
      acc = wmma_bf16(a1, ldBT(rhb + (size_t)j0*DHD + 32, DHD), acc);
#pragma unroll
      for (int r = 0; r < 8; ++r)
        raw[(rowsub*16 + r + 8*lh)*KL + j0 + nl] = acc[r];
    }
  }
  __syncthreads();

  // ---- phase 2: sc = (AC + rel_shift(raw)) * scale ----
  {
    const bf16* qub = qu + (size_t)bh * QLEN * DHD;
    const bf16* kb  = kk + (size_t)bh * KL * DHD;
    v16bf qa0 = ldA(qub + (size_t)i0*DHD, DHD);
    v16bf qa1 = ldA(qub + (size_t)i0*DHD + 32, DHD);
    for (int c = 0; c < 8; ++c) {
      int j0 = (wv * 8 + c) * 16;
      v8f acc = {};
      acc = wmma_bf16(qa0, ldBT(kb + (size_t)j0*DHD, DHD), acc);
      acc = wmma_bf16(qa1, ldBT(kb + (size_t)j0*DHD + 32, DHD), acc);
#pragma unroll
      for (int r = 0; r < 8; ++r) {
        int il = r + 8*lh;
        int i = i0 + il, j = j0 + nl;
        float bd;
        if (j <= i)          bd = raw[il*KL + (KL - 1 - (i - j))];
        else if (j == i + 1) bd = 0.f;
        else                 bd = raw[(il + 1)*KL + (j - i - 2)];
        sc[il*KL + j] = (acc[r] + bd) * 0.125f;
      }
    }
  }
  __syncthreads();

  // ---- phase 3: softmax over klen, write bf16 probs (overwrites raw) ----
  for (int rr = wv*2; rr < wv*2 + 2; ++rr) {
    float mx = -1e30f;
    for (int j = lane; j < KL; j += 32) mx = fmaxf(mx, sc[rr*KL + j]);
    for (int o = 16; o; o >>= 1) mx = fmaxf(mx, __shfl_xor(mx, o, 32));
    float sum = 0.f;
    for (int j = lane; j < KL; j += 32) {
      float e = __expf(sc[rr*KL + j] - mx);
      sc[rr*KL + j] = e; sum += e;
    }
    for (int o = 16; o; o >>= 1) sum += __shfl_xor(sum, o, 32);
    float inv = 1.f / sum;
    for (int j = lane; j < KL; j += 32) pbuf[rr*KL + j] = (bf16)(sc[rr*KL + j] * inv);
  }
  __syncthreads();

  // ---- phase 4: attn_vec = P(16x1024) x V(1024x64) ----
  {
    const bf16* vb = vvv + (size_t)bh * KL * DHD;
    int dt = wv & 3, kh = wv >> 2;
    v8f acc = {};
    for (int kt = 0; kt < 16; ++kt) {
      int kbase = kh*512 + kt*32;
      v16bf pa = ldA(pbuf + kbase, KL);
      v16bf bb = ldB(vb + (size_t)kbase*DHD + dt*16, DHD);
      acc = wmma_bf16(pa, bb, acc);
    }
#pragma unroll
    for (int r = 0; r < 8; ++r)
      part[(kh*16 + r + 8*lh)*64 + dt*16 + nl] = acc[r];
  }
  __syncthreads();
  for (int idx = threadIdx.x; idx < 16*64; idx += 256) {
    int rrow = idx >> 6, dh = idx & 63;
    float val = part[rrow*64 + dh] + part[(16 + rrow)*64 + dh];
    av[((size_t)(i0 + rrow)*BSZ + b)*DM + h*DHD + dh] = val;
  }
}

// ---------------- layernorm: out = LN(a+b)*g + beta ---------------------------

__global__ __launch_bounds__(256) void ln_kernel(
    const float* __restrict__ a, const float* __restrict__ b,
    const float* __restrict__ g, const float* __restrict__ be,
    float* __restrict__ of, bf16* __restrict__ ob)
{
  const int t = blockIdx.x;
  const size_t base = (size_t)t * DM;
  __shared__ float red[8];
  const int wv = threadIdx.x >> 5, lane = threadIdx.x & 31;
  float x[4]; float s = 0.f;
#pragma unroll
  for (int e = 0; e < 4; ++e) {
    int i = e*256 + threadIdx.x;
    x[e] = a[base + i] + b[base + i];
    s += x[e];
  }
  for (int o = 16; o; o >>= 1) s += __shfl_xor(s, o, 32);
  if (!lane) red[wv] = s;
  __syncthreads();
  float tot = red[0]+red[1]+red[2]+red[3]+red[4]+red[5]+red[6]+red[7];
  float mean = tot * (1.f/DM);
  __syncthreads();
  float vs = 0.f;
#pragma unroll
  for (int e = 0; e < 4; ++e) { float d = x[e] - mean; vs += d*d; }
  for (int o = 16; o; o >>= 1) vs += __shfl_xor(vs, o, 32);
  if (!lane) red[wv] = vs;
  __syncthreads();
  float vtot = red[0]+red[1]+red[2]+red[3]+red[4]+red[5]+red[6]+red[7];
  float rstd = rsqrtf(vtot * (1.f/DM) + 1e-5f);
#pragma unroll
  for (int e = 0; e < 4; ++e) {
    int i = e*256 + threadIdx.x;
    float y = (x[e] - mean) * rstd * g[i] + be[i];
    of[base + i] = y;
    if (ob) ob[base + i] = (bf16)y;
  }
}

// ---------------- misc kernels ------------------------------------------------

__global__ void cvt_kernel(const float* __restrict__ s, bf16* __restrict__ d, int n) {
  int i = blockIdx.x * blockDim.x + threadIdx.x;
  int stride = gridDim.x * blockDim.x;
  for (; i < n; i += stride) d[i] = (bf16)s[i];
}

__global__ void posemb_kernel(bf16* __restrict__ pe) {
  int idx = blockIdx.x * 256 + threadIdx.x;
  if (idx >= KL * DM) return;
  int r = idx >> 10, d = idx & 1023;
  float pos = (float)(KL - 1 - r);
  int f = (d < 512) ? d : d - 512;
  float freq = __expf(-(2.f * (float)f / (float)DM) * 9.210340371976184f); // ln(10000)
  float ang = pos * freq;
  pe[idx] = (bf16)((d < 512) ? sinf(ang) : cosf(ang));
}

__global__ void reshape_qkv(const float* __restrict__ wh, const float* __restrict__ u,
                            const float* __restrict__ v, bf16* __restrict__ qu,
                            bf16* __restrict__ qv, bf16* __restrict__ kk,
                            bf16* __restrict__ vvv) {
  int idx = blockIdx.x * 256 + threadIdx.x;
  if (idx >= BSZ*NH*QLEN*DHD) return;
  int dh = idx & 63, i = (idx >> 6) & 1023, h = (idx >> 16) & 15, b = idx >> 20;
  size_t t = (size_t)i * BSZ + b;
  int hd = h*DHD + dh;
  float q = wh[t*3*DM + hd];
  qu[idx]  = (bf16)(q + u[hd]);
  qv[idx]  = (bf16)(q + v[hd]);
  kk[idx]  = (bf16)wh[t*3*DM + DM + hd];
  vvv[idx] = (bf16)wh[t*3*DM + 2*DM + hd];
}

__global__ void reshape_rh(const float* __restrict__ rt, bf16* __restrict__ rh) {
  int idx = blockIdx.x * 256 + threadIdx.x;
  if (idx >= NH*KL*DHD) return;
  int dh = idx & 63, j = (idx >> 6) & 1023, h = idx >> 16;
  rh[idx] = (bf16)rt[(size_t)j*DM + h*DHD + dh];
}

// ---------------- host orchestration -----------------------------------------

extern "C" void kernel_launch(void* const* d_in, const int* in_sizes, int n_in,
                              void* d_out, int out_size, void* d_ws, size_t ws_size,
                              hipStream_t stream) {
  (void)in_sizes; (void)n_in; (void)out_size; (void)ws_size;
  const float* in_x  = (const float*)d_in[0];
  const float* qkv_w = (const float*)d_in[1];
  const float* r_w   = (const float*)d_in[2];
  const float* o_w   = (const float*)d_in[3];
  const float* ln1_g = (const float*)d_in[4];
  const float* ln1_b = (const float*)d_in[5];
  const float* ln2_g = (const float*)d_in[6];
  const float* ln2_b = (const float*)d_in[7];
  const float* w1    = (const float*)d_in[8];
  const float* b1    = (const float*)d_in[9];
  const float* w2    = (const float*)d_in[10];
  const float* b2    = (const float*)d_in[11];
  const float* u     = (const float*)d_in[12];
  const float* v     = (const float*)d_in[13];
  const float* out_w = (const float*)d_in[14];

  char* ws = (char*)d_ws; size_t off = 0;
  auto alloc = [&](size_t bytes) -> void* {
    void* p = ws + off; off += (bytes + 255) & ~(size_t)255; return p;
  };

  bf16* qkvw_b = (bf16*)alloc((size_t)NL*3*DM*DM*2);
  bf16* rw_b   = (bf16*)alloc((size_t)NL*DM*DM*2);
  bf16* ow_b   = (bf16*)alloc((size_t)NL*DM*DM*2);
  bf16* w1_b   = (bf16*)alloc((size_t)NL*FF*DM*2);
  bf16* w2_b   = (bf16*)alloc((size_t)NL*DM*FF*2);
  bf16* outw_b = (bf16*)alloc((size_t)OUTD*DM*2);
  bf16* pe_b   = (bf16*)alloc((size_t)KL*DM*2);
  float* core   = (float*)alloc((size_t)NT*DM*4);
  bf16*  core_b = (bf16*)alloc((size_t)NT*DM*2);
  float* wheads = (float*)alloc((size_t)NT*3*DM*4);
  float* rtmp   = (float*)alloc((size_t)KL*DM*4);
  bf16* qu_b = (bf16*)alloc((size_t)BSZ*NH*QLEN*DHD*2);
  bf16* qv_b = (bf16*)alloc((size_t)BSZ*NH*QLEN*DHD*2);
  bf16* kk_b = (bf16*)alloc((size_t)BSZ*NH*KL*DHD*2);
  bf16* vv_b = (bf16*)alloc((size_t)BSZ*NH*KL*DHD*2);
  bf16* rh_b = (bf16*)alloc((size_t)NH*KL*DHD*2);
  float* attnvec   = (float*)alloc((size_t)NT*DM*4);
  bf16*  attnvec_b = (bf16*)alloc((size_t)NT*DM*2);
  float* ao   = (float*)alloc((size_t)NT*DM*4);
  float* yb   = (float*)alloc((size_t)NT*DM*4);
  bf16*  yb_b = (bf16*)alloc((size_t)NT*DM*2);
  bf16*  h1_b = (bf16*)alloc((size_t)NT*FF*2);
  float* h2   = (float*)alloc((size_t)NT*DM*4);

  auto cvt = [&](const float* s, bf16* d, size_t n) {
    int blocks = (int)((n + 2047) / 2048);
    cvt_kernel<<<blocks, 256, 0, stream>>>(s, d, (int)n);
  };
  auto gemm = [&](const bf16* A, const bf16* Wt, const float* bias,
                  float* Cf, bf16* Cb, int M, int N, int K, int relu) {
    dim3 g(N/128, M/128);
    gemm_kernel<<<g, 256, 0, stream>>>(A, Wt, bias, Cf, Cb, M, N, K, relu);
  };

  // weight + input conversion
  cvt(qkv_w, qkvw_b, (size_t)NL*3*DM*DM);
  cvt(r_w,   rw_b,   (size_t)NL*DM*DM);
  cvt(o_w,   ow_b,   (size_t)NL*DM*DM);
  cvt(w1,    w1_b,   (size_t)NL*FF*DM);
  cvt(w2,    w2_b,   (size_t)NL*DM*FF);
  cvt(out_w, outw_b, (size_t)OUTD*DM);
  cvt(in_x,  core_b, (size_t)NT*DM);
  posemb_kernel<<<(KL*DM)/256, 256, 0, stream>>>(pe_b);
  hipMemcpyAsync(core, in_x, (size_t)NT*DM*4, hipMemcpyDeviceToDevice, stream);

  float* outp = (float*)d_out;
  const size_t OUT_ELEMS = (size_t)NT*OUTD;
  const size_t MEMSLICE  = (size_t)MEMLEN*BSZ*DM;   // 2,097,152 floats

  // hids[0] slice (rows 512..1023 of input)
  hipMemcpyAsync(outp + OUT_ELEMS, in_x + MEMSLICE, MEMSLICE*4,
                 hipMemcpyDeviceToDevice, stream);

  for (int l = 0; l < NL; ++l) {
    gemm(core_b, qkvw_b + (size_t)l*3*DM*DM, nullptr, wheads, nullptr, NT, 3*DM, DM, 0);
    gemm(pe_b,   rw_b   + (size_t)l*DM*DM,   nullptr, rtmp,   nullptr, KL, DM, DM, 0);
    reshape_qkv<<<(BSZ*NH*QLEN*DHD)/256, 256, 0, stream>>>(wheads, u, v, qu_b, qv_b, kk_b, vv_b);
    reshape_rh<<<(NH*KL*DHD)/256, 256, 0, stream>>>(rtmp, rh_b);
    attn_kernel<<<dim3(QLEN/16, BSZ*NH), 256, 196608, stream>>>(qu_b, qv_b, kk_b, vv_b, rh_b, attnvec);
    cvt(attnvec, attnvec_b, (size_t)NT*DM);
    gemm(attnvec_b, ow_b + (size_t)l*DM*DM, nullptr, ao, nullptr, NT, DM, DM, 0);
    ln_kernel<<<NT, 256, 0, stream>>>(core, ao, ln1_g + l*DM, ln1_b + l*DM, yb, yb_b);
    gemm(yb_b, w1_b + (size_t)l*FF*DM, b1 + (size_t)l*FF, nullptr, h1_b, NT, FF, DM, 1);
    gemm(h1_b, w2_b + (size_t)l*DM*FF, b2 + (size_t)l*DM, h2, nullptr, NT, DM, FF, 0);
    ln_kernel<<<NT, 256, 0, stream>>>(yb, h2, ln2_g + l*DM, ln2_b + l*DM, core, core_b);
    hipMemcpyAsync(outp + OUT_ELEMS + (size_t)(l+1)*MEMSLICE, core + MEMSLICE,
                   MEMSLICE*4, hipMemcpyDeviceToDevice, stream);
  }

  // final projection: out = core @ out_w^T  -> d_out[0 .. NT*OUTD)
  gemm(core_b, outw_b, nullptr, outp, nullptr, NT, OUTD, DM, 0);
}
